// GATActor_26723286516188
// MI455X (gfx1250) — matrix operand order repaired and checked by reading, not compiled, not run on hardware
//
#include <hip/hip_runtime.h>

// ---------------- problem constants ----------------
constexpr int NAG   = 5;
constexpr int Bgr   = 512;
constexpr int NP    = 50;
constexpr int Nn    = Bgr * NP;       // 25600 nodes
constexpr int EPG   = 400;
constexpr int Ee    = Bgr * EPG;      // 204800 edges
constexpr int Hh    = 8;
constexpr int Cc    = 64;
constexpr int Ff    = Hh * Cc;        // 512
constexpr int E2    = Ee + Nn;        // edges + self loops
constexpr int GRUI  = 2 * Ff;         // 1024
constexpr int GRUH  = 128;
constexpr int G3    = 3 * GRUH;       // 384

// ---------------- vector types ----------------
typedef __attribute__((ext_vector_type(16))) __bf16        bf16x16;
typedef __attribute__((ext_vector_type(8)))  float         floatx8;
typedef __attribute__((ext_vector_type(2)))  float         floatx2;
typedef __attribute__((ext_vector_type(4)))  unsigned int  uint4v;
typedef __attribute__((ext_vector_type(4)))  unsigned int  u32x4;
typedef __attribute__((ext_vector_type(8)))  int           i32x8;
typedef __attribute__((ext_vector_type(4)))  int           i32x4;

// fp32 -> bf16 bits, round-to-nearest-even
__device__ __forceinline__ unsigned short f2bf(float f) {
  unsigned u = __float_as_uint(f);
  unsigned r = u + 0x7fffu + ((u >> 16) & 1u);
  return (unsigned short)(r >> 16);
}
// order-preserving float<->uint mapping (atomic max over floats incl. negatives)
__device__ __forceinline__ unsigned ordKey(float f) {
  unsigned u = __float_as_uint(f);
  return (u & 0x80000000u) ? ~u : (u | 0x80000000u);
}
__device__ __forceinline__ float ordVal(unsigned k) {
  unsigned u = (k & 0x80000000u) ? (k & 0x7fffffffu) : ~k;
  return __uint_as_float(u);
}
__device__ __forceinline__ float sigm(float x) { return 1.f / (1.f + __expf(-x)); }

#if defined(__AMDGCN__) && __has_builtin(__builtin_amdgcn_tensor_load_to_lds)
// Issue a TDM 2-D tile load: `rows` consecutive rows of length K (bf16) from
// `gsrc` (row-major, stride K) into LDS at byte offset ldsOff.
// Toolchain is the 6-arg (clang-23 / therock) form:
//   (uint32x4 g0, int32x8 g1, int32x4 g2, int32x4 g3, int32x8 g4, i32 cpol)
__device__ __forceinline__ void tdm_load_panel(const unsigned short* gsrc, unsigned ldsOff,
                                               int K, int rows) {
  unsigned long long ga = (unsigned long long)(const void*)gsrc;
  u32x4 g0;
  g0[0] = 1u;                                         // count=1, user descriptor
  g0[1] = ldsOff;                                     // lds_addr
  g0[2] = (unsigned)ga;                               // global_addr[31:0]
  g0[3] = (unsigned)((ga >> 32) & 0x1ffffffu) | (2u << 30);  // addr[56:32] | type=2
  i32x8 g1;
  unsigned td0 = (unsigned)K;                         // tensor_dim0 = K
  unsigned td1 = 0x100000u;                           // tensor_dim1 (OOB bound, generous)
  g1[0] = (int)(1u << 16);                            // data_size=1 (2 bytes)
  g1[1] = (int)((td0 & 0xffffu) << 16);               // tensor_dim0[15:0]
  g1[2] = (int)(((td0 >> 16) & 0xffffu) | ((td1 & 0xffffu) << 16));
  g1[3] = (int)(((td1 >> 16) & 0xffffu) | (((unsigned)K & 0xffffu) << 16)); // tile_dim0=K
  g1[4] = (int)((unsigned)rows & 0xffffu);            // tile_dim1 = rows
  g1[5] = (int)td0;                                   // tensor_dim0_stride = K
  g1[6] = 0;
  g1[7] = 0;
  i32x4 gz4 = (i32x4)(0);
  i32x8 gz8 = (i32x8)(0);
  __builtin_amdgcn_tensor_load_to_lds(g0, g1, gz4, gz4, gz8, 0);
}
#endif

// ---------------- conv1: [Nn,4] @ [4,Ff] via V_WMMA_F32_16X16X4_F32 ----------------
__global__ void conv1_gemm_k(const float* __restrict__ X, const float* __restrict__ W1,
                             float* __restrict__ Hout) {
  int gw   = (blockIdx.x * blockDim.x + threadIdx.x) >> 5;
  int lane = threadIdx.x & 31;
  const int ntiles = Ff / 16;                // 32
  int tm = gw / ntiles, tn = gw - tm * ntiles;
  if (tm >= Nn / 16) return;
  bool hi = lane >= 16;
  int l15 = lane & 15;
  int mrow = tm * 16 + l15;
  int ncol = tn * 16 + l15;
#if __has_builtin(__builtin_amdgcn_wmma_f32_16x16x4_f32)
  int ka = hi ? 2 : 0;
  floatx2 a, b;
  a.x = X[mrow * 4 + ka];     a.y = X[mrow * 4 + ka + 1];
  b.x = W1[ka * Ff + ncol];   b.y = W1[(ka + 1) * Ff + ncol];
  floatx8 acc = {};
  acc = __builtin_amdgcn_wmma_f32_16x16x4_f32(false, a, false, b, (short)0, acc, false, false);
  for (int r = 0; r < 8; ++r)
    Hout[(size_t)(tm * 16 + r + (hi ? 8 : 0)) * Ff + ncol] = acc[r];
#else
  for (int r = 0; r < 8; ++r) {
    int rr = tm * 16 + r + (hi ? 8 : 0);
    float s = 0.f;
    for (int k = 0; k < 4; ++k) s += X[rr * 4 + k] * W1[k * Ff + ncol];
    Hout[(size_t)rr * Ff + ncol] = s;
  }
#endif
}

// ---------------- bf16 conversion / transposed-weight prep ----------------
__global__ void f32_to_bf16_k(const float* __restrict__ s, unsigned short* __restrict__ d, int n) {
  int i = blockIdx.x * blockDim.x + threadIdx.x;
  if (i < n) d[i] = f2bf(s[i]);
}
__global__ void transpose_bf16_k(const float* __restrict__ W, unsigned short* __restrict__ Wt,
                                 int rows, int cols) {
  int i = blockIdx.x * blockDim.x + threadIdx.x;
  if (i < rows * cols) {
    int r = i / cols, c = i - r * cols;
    Wt[(size_t)c * rows + r] = f2bf(W[i]);
  }
}

// ---------------- main GEMM: C[M,N] = A(bf16,[M,K]) @ Bt(bf16,[N,K])^T ----------------
// Block = 256 threads (8 waves). Block tile = 128 rows x 64 cols; wave w -> rows
// [w*16,w*16+16) x all 64 cols (4 WMMA accumulators, A fragment reused 4x).
// B panel (64 x K) and optionally A panel (128 x K) staged in LDS by the TDM.
__global__ void gemm_bf16_wmma_k(const unsigned short* __restrict__ A,
                                 const unsigned short* __restrict__ Bt,
                                 float* __restrict__ Cm, int M, int Ncols, int K, int stageA) {
  extern __shared__ unsigned short lds[];
  unsigned short* Bs = lds;                        // [64][K]
  unsigned short* As = lds + (size_t)64 * K;       // [128][K] when stageA
  const int tid = threadIdx.x;
  const int wv = tid >> 5, lane = tid & 31;
  const bool hi = lane >= 16;
  const int l15 = lane & 15;
  const int nColBlk = Ncols >> 6;
  const int mb = blockIdx.x / nColBlk;
  const int colBase = (blockIdx.x - mb * nColBlk) << 6;
  const int rowBase = mb * 128;

#if defined(__AMDGCN__) && __has_builtin(__builtin_amdgcn_tensor_load_to_lds)
  if (wv == 0) {
    unsigned ldsBase = __builtin_amdgcn_groupstaticsize();   // dynamic-LDS start
    tdm_load_panel(Bt + (size_t)colBase * K, ldsBase, K, 64);
    if (stageA)
      tdm_load_panel(A + (size_t)rowBase * K, ldsBase + 64u * (unsigned)K * 2u, K, 128);
    __builtin_amdgcn_s_wait_tensorcnt(0);
  }
#else
  for (int i = tid; i < 64 * K / 8; i += blockDim.x)
    ((uint4v*)Bs)[i] = ((const uint4v*)(Bt + (size_t)colBase * K))[i];
  if (stageA)
    for (int i = tid; i < 128 * K / 8; i += blockDim.x)
      ((uint4v*)As)[i] = ((const uint4v*)(A + (size_t)rowBase * K))[i];
#endif
  __syncthreads();

  const unsigned short* arow = stageA
      ? (As + (size_t)(wv * 16 + l15) * K)
      : (A + (size_t)(rowBase + wv * 16 + l15) * K);
  floatx8 ac[4] = {};
  for (int k0 = 0; k0 < K; k0 += 32) {
    // A 16x32 bf16 fragment: lanes0-15 elems = K {k0..+7, k0+16..+23}; lanes16-31 +8
    union { uint4v q[2]; bf16x16 v; } af;
    int a0 = k0 + (hi ? 8 : 0);
    af.q[0] = *(const uint4v*)(arow + a0);
    af.q[1] = *(const uint4v*)(arow + a0 + 16);
    int b0 = k0 + (hi ? 8 : 0);
#pragma unroll
    for (int j = 0; j < 4; ++j) {
      // B 32x16 fragment: col = j*16 + lane&15; lanes0-15 K {k0..+7, k0+16..+23}; lanes16-31 +8
      union { uint4v q[2]; bf16x16 v; } bfr;
      const unsigned short* bp = Bs + (size_t)(j * 16 + l15) * K + b0;
      bfr.q[0] = *(const uint4v*)bp;
      bfr.q[1] = *(const uint4v*)(bp + 16);
      ac[j] = __builtin_amdgcn_wmma_f32_16x16x32_bf16(false, af.v, false, bfr.v,
                                                      (short)0, ac[j], false, false);
    }
  }
#pragma unroll
  for (int j = 0; j < 4; ++j) {
    int c = colBase + j * 16 + l15;
    for (int r = 0; r < 8; ++r)
      Cm[(size_t)(rowBase + wv * 16 + r + (hi ? 8 : 0)) * Ncols + c] = ac[j][r];
  }
}

// ---------------- attention dot products: s_src/s_dst [Nn,Hh] ----------------
__global__ void attn_dots_k(const float* __restrict__ Hm, const float* __restrict__ asrc,
                            const float* __restrict__ adst, float* __restrict__ ssrc,
                            float* __restrict__ sdst) {
  int idx = blockIdx.x * blockDim.x + threadIdx.x;
  if (idx >= Nn * Hh) return;
  int n = idx >> 3, h = idx & 7;
  const float* hp = Hm + (size_t)n * Ff + h * Cc;
  float s1 = 0.f, s2 = 0.f;
  for (int c = 0; c < Cc; ++c) { s1 += hp[c] * asrc[h * Cc + c]; s2 += hp[c] * adst[h * Cc + c]; }
  ssrc[idx] = s1; sdst[idx] = s2;
}

__global__ void weae_k(const float* __restrict__ We, const float* __restrict__ ae,
                       float* __restrict__ weae) {
  int h = threadIdx.x;
  if (h < Hh) {
    float s = 0.f;
    for (int c = 0; c < Cc; ++c) s += We[h * Cc + c] * ae[h * Cc + c];
    weae[h] = s;
  }
}

// ---------------- per-node mean incoming edge_attr (self-loop fill value) ----------------
__global__ void loop_attr_k(const int* __restrict__ dstA, const float* __restrict__ ea,
                            float* __restrict__ loopA) {
  __shared__ float ssum[NP];
  __shared__ float scnt[NP];
  int g = blockIdx.x, tid = threadIdx.x;
  if (tid < NP) { ssum[tid] = 0.f; scnt[tid] = 0.f; }
  __syncthreads();
  for (int e = tid; e < EPG; e += blockDim.x) {
    int ge = g * EPG + e;
    int dl = dstA[ge] - g * NP;
    atomicAdd(&ssum[dl], ea[ge]);
    atomicAdd(&scnt[dl], 1.f);
  }
  __syncthreads();
  if (tid < NP) loopA[g * NP + tid] = ssum[tid] / fmaxf(scnt[tid], 1.f);
}

// ---------------- GAT edge softmax + aggregation, one block per graph ----------------
__global__ void gat_layer_k(const float* __restrict__ Hm, const float* __restrict__ ssrc,
                            const float* __restrict__ sdst, const int* __restrict__ srcA,
                            const int* __restrict__ dstA, const float* __restrict__ ea,
                            const float* __restrict__ loopA, const float* __restrict__ weae,
                            const float* __restrict__ bias, float* __restrict__ pbuf,
                            float* __restrict__ outX, int withLoops) {
  extern __shared__ float smem[];
  float*    outAcc = smem;
  unsigned* amaxU  = (unsigned*)(smem + NP * Ff);
  float*    denom  = smem + NP * Ff + NP * Hh;

  const int g = blockIdx.x, tid = threadIdx.x, bd = blockDim.x;
  const int nE = EPG + (withLoops ? NP : 0);

  for (int t = tid; t < NP * Ff; t += bd) outAcc[t] = 0.f;
  for (int t = tid; t < NP * Hh; t += bd) { amaxU[t] = 0u; denom[t] = 0.f; }
  __syncthreads();

  for (int t = tid; t < nE * Hh; t += bd) {          // pass 1: segment max
    int e = t >> 3, h = t & 7, s, d; float attr;
    if (e < EPG) { int ge = g * EPG + e; s = srcA[ge]; d = dstA[ge]; attr = ea[ge]; }
    else         { int nd = g * NP + (e - EPG); s = nd; d = nd; attr = loopA[nd]; }
    float al = ssrc[s * Hh + h] + sdst[d * Hh + h] + attr * weae[h];
    al = al > 0.f ? al : 0.2f * al;
    atomicMax(&amaxU[(d - g * NP) * Hh + h], ordKey(al));
  }
  __syncthreads();

  for (int t = tid; t < nE * Hh; t += bd) {          // pass 2: exp + segment sum
    int e = t >> 3, h = t & 7, s, d, prow; float attr;
    if (e < EPG) { int ge = g * EPG + e; s = srcA[ge]; d = dstA[ge]; attr = ea[ge]; prow = ge; }
    else         { int nd = g * NP + (e - EPG); s = nd; d = nd; attr = loopA[nd]; prow = Ee + nd; }
    float al = ssrc[s * Hh + h] + sdst[d * Hh + h] + attr * weae[h];
    al = al > 0.f ? al : 0.2f * al;
    int dl = d - g * NP;
    float p = __expf(al - ordVal(amaxU[dl * Hh + h]));
    pbuf[(size_t)prow * Hh + h] = p;
    atomicAdd(&denom[dl * Hh + h], p);
  }
  __syncthreads();

  for (int t = tid; t < nE * Hh; t += bd) {          // pass 3: weighted aggregation
    int e = t >> 3, h = t & 7, s, d, prow;
    if (e < EPG) { int ge = g * EPG + e; s = srcA[ge]; d = dstA[ge]; prow = ge; }
    else         { int nd = g * NP + (e - EPG); s = nd; d = nd; prow = Ee + nd; }
    int dl = d - g * NP;
    float alpha = pbuf[(size_t)prow * Hh + h] / (denom[dl * Hh + h] + 1e-16f);
    const float* hs = Hm + (size_t)s * Ff + h * Cc;
    float* oa = outAcc + dl * Ff + h * Cc;
    for (int c = 0; c < Cc; ++c) atomicAdd(&oa[c], alpha * hs[c]);
  }
  __syncthreads();

  for (int t = tid; t < NP * Ff; t += bd) {          // write: relu(out + bias)
    int n = t / Ff, f = t - n * Ff;
    float v = outAcc[t] + bias[f];
    outX[(size_t)(g * NP + n) * Ff + f] = v > 0.f ? v : 0.f;
  }
}

// ---------------- head: pooling / combined / GRU / FC ----------------
__global__ void pool_k(const float* __restrict__ x3, float* __restrict__ gemb) {
  int idx = blockIdx.x * blockDim.x + threadIdx.x;
  if (idx >= Bgr * Ff) return;
  int g = idx / Ff, f = idx - g * Ff;
  float s = 0.f;
  for (int n = 0; n < NP; ++n) s += x3[(size_t)(g * NP + n) * Ff + f];
  gemb[idx] = s * (1.f / NP);
}

__global__ void build_combined_k(const float* __restrict__ x3, const float* __restrict__ gemb,
                                 unsigned short* __restrict__ comb) {
  int idx = blockIdx.x * blockDim.x + threadIdx.x;
  if (idx >= Bgr * NAG * GRUI) return;
  int row = idx / GRUI, col = idx - row * GRUI;
  int g = row / NAG, a = row - g * NAG;
  float v = (col < Ff) ? x3[(size_t)(g * NP + a) * Ff + col] : gemb[g * Ff + (col - Ff)];
  comb[idx] = f2bf(v);
}

__global__ void gru_step_k(const float* __restrict__ giAll, const float* __restrict__ Whh,
                           const float* __restrict__ bih, const float* __restrict__ bhh,
                           float* __restrict__ hbuf, float* __restrict__ gruout, int t) {
  __shared__ float hs[GRUH];
  __shared__ float gh[G3];
  int g = blockIdx.x, j = threadIdx.x;     // blockDim = 128
  hs[j] = hbuf[g * GRUH + j];
  __syncthreads();
  float g0 = 0.f, g1 = 0.f, g2 = 0.f;
  for (int k = 0; k < GRUH; ++k) {
    float hv = hs[k];
    g0 += hv * Whh[k * G3 + j];
    g1 += hv * Whh[k * G3 + j + GRUH];
    g2 += hv * Whh[k * G3 + j + 2 * GRUH];
  }
  gh[j] = g0; gh[j + GRUH] = g1; gh[j + 2 * GRUH] = g2;
  __syncthreads();
  const float* gi = giAll + (size_t)(g * NAG + t) * G3;
  float ir = gi[j] + bih[j];
  float iz = gi[j + GRUH] + bih[j + GRUH];
  float in = gi[j + 2 * GRUH] + bih[j + 2 * GRUH];
  float hr = gh[j] + bhh[j];
  float hz = gh[j + GRUH] + bhh[j + GRUH];
  float hn = gh[j + 2 * GRUH] + bhh[j + 2 * GRUH];
  float r = sigm(ir + hr);
  float z = sigm(iz + hz);
  float nn = tanhf(in + r * hn);
  float hnew = (1.f - z) * nn + z * hs[j];
  hbuf[g * GRUH + j] = hnew;
  gruout[(size_t)(g * NAG + t) * GRUH + j] = hnew;
}

__global__ void fc1_k(const float* __restrict__ go, const float* __restrict__ W,
                      const float* __restrict__ b, float* __restrict__ f1) {
  int idx = blockIdx.x * blockDim.x + threadIdx.x;
  if (idx >= Bgr * NAG * 64) return;
  int row = idx >> 6, o = idx & 63;
  float s = b[o];
  for (int k = 0; k < GRUH; ++k) s += go[(size_t)row * GRUH + k] * W[k * 64 + o];
  f1[idx] = s > 0.f ? s : 0.f;
}

__global__ void fc2_k(const float* __restrict__ f1, const float* __restrict__ W,
                      const float* __restrict__ b, const float* __restrict__ log_std,
                      float* __restrict__ out) {
  int idx = blockIdx.x * blockDim.x + threadIdx.x;
  if (idx >= Bgr * NAG * 3) return;
  int row = idx / 3, o = idx - row * 3;
  float s = b[o];
  for (int k = 0; k < 64; ++k) s += f1[row * 64 + k] * W[k * 3 + o];
  out[idx] = s;                                            // action_mean
  float ls = fminf(fmaxf(log_std[o], -20.f), 2.f);
  out[Bgr * NAG * 3 + idx] = __expf(ls);                   // action_std
}

__global__ void hlast_k(const float* __restrict__ h, float* __restrict__ out) {
  int idx = blockIdx.x * blockDim.x + threadIdx.x;
  if (idx < Bgr * GRUH) out[2 * Bgr * NAG * 3 + idx] = h[idx];
}

// ---------------- host orchestration ----------------
extern "C" void kernel_launch(void* const* d_in, const int* in_sizes, int n_in,
                              void* d_out, int out_size, void* d_ws, size_t ws_size,
                              hipStream_t stream) {
  const float* x      = (const float*)d_in[0];
  const int*   ei     = (const int*)d_in[1];
  const int*   srcA   = ei;
  const int*   dstA   = ei + Ee;
  const float* ea     = (const float*)d_in[2];
  const float* hidden = (const float*)d_in[3];
  const float* W[3]    = {(const float*)d_in[4],  (const float*)d_in[10], (const float*)d_in[16]};
  const float* aS[3]   = {(const float*)d_in[5],  (const float*)d_in[11], (const float*)d_in[17]};
  const float* aD[3]   = {(const float*)d_in[6],  (const float*)d_in[12], (const float*)d_in[18]};
  const float* We[3]   = {(const float*)d_in[7],  (const float*)d_in[13], (const float*)d_in[19]};
  const float* aE[3]   = {(const float*)d_in[8],  (const float*)d_in[14], (const float*)d_in[20]};
  const float* bb[3]   = {(const float*)d_in[9],  (const float*)d_in[15], (const float*)d_in[21]};
  const float* gWih = (const float*)d_in[22];
  const float* gWhh = (const float*)d_in[23];
  const float* gbih = (const float*)d_in[24];
  const float* gbhh = (const float*)d_in[25];
  const float* f1W  = (const float*)d_in[26];
  const float* f1b  = (const float*)d_in[27];
  const float* f2W  = (const float*)d_in[28];
  const float* f2b  = (const float*)d_in[29];
  const float* lstd = (const float*)d_in[30];
  float* out = (float*)d_out;

  // workspace carve-up (~200 MB total)
  char* w = (char*)d_ws;
  size_t off = 0;
  auto take = [&](size_t bytes) -> void* {
    void* p = w + off;
    off = (off + bytes + 255) & ~(size_t)255;
    return p;
  };
  float*          bufX  = (float*)take((size_t)Nn * Ff * 4);
  float*          bufY  = (float*)take((size_t)Nn * Ff * 4);
  float*          bufH  = (float*)take((size_t)Nn * Ff * 4);
  unsigned short* aBf   = (unsigned short*)take((size_t)Nn * Ff * 2);
  unsigned short* wtBf  = (unsigned short*)take((size_t)GRUI * G3 * 2);
  float*          ssrc  = (float*)take((size_t)Nn * Hh * 4);
  float*          sdst  = (float*)take((size_t)Nn * Hh * 4);
  float*          pbuf  = (float*)take((size_t)E2 * Hh * 4);
  float*          loopA = (float*)take((size_t)Nn * 4);
  float*          weae  = (float*)take(Hh * 4);
  float*          giAll = (float*)take((size_t)Bgr * NAG * G3 * 4);
  float*          hstat = (float*)take((size_t)Bgr * GRUH * 4);
  float*          gout  = (float*)take((size_t)Bgr * NAG * GRUH * 4);
  float*          f1buf = (float*)take((size_t)Bgr * NAG * 64 * 4);
  float*          gemb  = (float*)take((size_t)Bgr * Ff * 4);

  auto cdiv = [](int a, int b) { return (a + b - 1) / b; };
  const int gatSmem = (NP * Ff + 2 * NP * Hh) * 4;            // 105,600 B LDS
  // main GEMM: 128x64 block tiles, A+B panels staged in LDS (192 KB)
  const int gemmBlocksMain = (Nn / 128) * (Ff / 64);          // 1600
  const int gemmSmemMain   = (64 * Ff + 128 * Ff) * 2;        // 196,608 B
  const int gemmBlocksGru  = (Bgr * NAG / 128) * (G3 / 64);   // 120
  const int gemmSmemGru    = 64 * GRUI * 2;                   // 131,072 B (B panel only)

  // ---- conv1 (fp32 WMMA, K=4) ----
  conv1_gemm_k<<<cdiv((Nn / 16) * (Ff / 16), 8), 256, 0, stream>>>(x, W[0], bufH);
  attn_dots_k<<<cdiv(Nn * Hh, 256), 256, 0, stream>>>(bufH, aS[0], aD[0], ssrc, sdst);
  weae_k<<<1, 64, 0, stream>>>(We[0], aE[0], weae);
  gat_layer_k<<<Bgr, 256, gatSmem, stream>>>(bufH, ssrc, sdst, srcA, dstA, ea, loopA, weae,
                                             bb[0], pbuf, bufX, 0);

  loop_attr_k<<<Bgr, 256, 0, stream>>>(dstA, ea, loopA);

  // ---- conv2 / conv3 (bf16 WMMA, K=512, TDM-staged panels) ----
  float* lin[2]  = {bufX, bufY};
  float* lout[2] = {bufY, bufX};
  for (int L = 0; L < 2; ++L) {
    int li = L + 1;
    f32_to_bf16_k<<<cdiv(Nn * Ff, 256), 256, 0, stream>>>(lin[L], aBf, Nn * Ff);
    transpose_bf16_k<<<cdiv(Ff * Ff, 256), 256, 0, stream>>>(W[li], wtBf, Ff, Ff);
    gemm_bf16_wmma_k<<<gemmBlocksMain, 256, gemmSmemMain, stream>>>(aBf, wtBf, bufH,
                                                                    Nn, Ff, Ff, 1);
    attn_dots_k<<<cdiv(Nn * Hh, 256), 256, 0, stream>>>(bufH, aS[li], aD[li], ssrc, sdst);
    weae_k<<<1, 64, 0, stream>>>(We[li], aE[li], weae);
    gat_layer_k<<<Bgr, 256, gatSmem, stream>>>(bufH, ssrc, sdst, srcA, dstA, ea, loopA, weae,
                                               bb[li], pbuf, lout[L], 1);
  }
  // x3 now in bufX

  // ---- head ----
  pool_k<<<cdiv(Bgr * Ff, 256), 256, 0, stream>>>(bufX, gemb);
  build_combined_k<<<cdiv(Bgr * NAG * GRUI, 256), 256, 0, stream>>>(bufX, gemb, aBf);
  transpose_bf16_k<<<cdiv(GRUI * G3, 256), 256, 0, stream>>>(gWih, wtBf, GRUI, G3);
  gemm_bf16_wmma_k<<<gemmBlocksGru, 256, gemmSmemGru, stream>>>(aBf, wtBf, giAll,
                                                                Bgr * NAG, G3, GRUI, 0);

  (void)hipMemcpyAsync(hstat, hidden, (size_t)Bgr * GRUH * 4,
                       hipMemcpyDeviceToDevice, stream);
  for (int t = 0; t < NAG; ++t)
    gru_step_k<<<Bgr, GRUH, 0, stream>>>(giAll, gWhh, gbih, gbhh, hstat, gout, t);

  fc1_k<<<cdiv(Bgr * NAG * 64, 256), 256, 0, stream>>>(gout, f1W, f1b, f1buf);
  fc2_k<<<cdiv(Bgr * NAG * 3, 256), 256, 0, stream>>>(f1buf, f2W, f2b, lstd, out);
  hlast_k<<<cdiv(Bgr * GRUH, 256), 256, 0, stream>>>(hstat, out);
}
